// StructuredOut_23141283790925
// MI455X (gfx1250) — matrix operand order repaired and checked by reading, not compiled, not run on hardware
//
#include <hip/hip_runtime.h>
#include <hip/hip_bf16.h>

// ---------------------------------------------------------------------------
// Problem constants (from reference): B=64, S=1024, D=768, NT=14, T=4
// Emission GEMM: (B*S=65536, D=768) x (D, NT*T=56) -> (B,NT,S,T)
// Then Viterbi over N=B*NT=896 sequences, S=1024 steps, T=4 tags.
// ---------------------------------------------------------------------------
#define B_      64
#define S_      1024
#define D_      768
#define NT_     14
#define T_      4
#define KC_     56            // valid output columns
#define KTILES_ 24            // 768 / 32 (bf16 WMMA K=32)
#define MROWS_  (B_ * S_)     // 65536
#define NSEQ_   (B_ * NT_)    // 896

// W-fragment workspace: per k-tile one contiguous 8KB slab:
//   [hi: 4 ntiles x 32 lanes x 16 bf16 = 2048 elems][lo: 2048 elems]
#define SLAB_ELEMS  4096                       // bf16 per k-tile slab (8192 B)
#define WFR_ELEMS   (KTILES_ * SLAB_ELEMS)     // 98304 bf16 = 196608 B
#define BP_OFF_BYTES (WFR_ELEMS * 2)           // backpointers after W frags

typedef __attribute__((ext_vector_type(16))) __bf16 v16bf;
typedef __attribute__((ext_vector_type(8)))  float  v8f;
typedef __attribute__((ext_vector_type(4)))  unsigned int u32x4_t;
typedef __attribute__((ext_vector_type(8)))  int i32x8_t;
typedef __attribute__((ext_vector_type(4)))  int i32x4_t;

// ---------------------------------------------------------------------------
// TDM: 1-D async copy of one 8192-byte k-tile slab (global -> LDS).
// D# per CDNA5 ISA ch.8: group0 = {count, lds_addr, global_addr, type=2},
// group1 = {data_size=8B, tensor_dim0=1024, tensor_dim1=1, tile_dim0=1024,
//           tile_dim1=1, tensor_dim0_stride=1024}; groups 2/3 = 0 (<=2D).
// This toolchain exposes the 6-arg builtin (extra zero-filled i32x8 group).
// ---------------------------------------------------------------------------
__device__ __forceinline__ void tdm_load_slab(const void* gsrc, unsigned lds_off) {
  unsigned long long ga = (unsigned long long)(uintptr_t)gsrc;
  u32x4_t g0;
  g0[0] = 1u;                                           // count=1 (valid user D#)
  g0[1] = lds_off;                                      // lds_addr (bytes)
  g0[2] = (unsigned)ga;                                 // global_addr[31:0]
  g0[3] = (unsigned)((ga >> 32) & 0x01FFFFFFull)        // global_addr[56:32]
          | 0x80000000u;                                // type=2 ("image")
  i32x8_t g1;
  g1[0] = (int)(3u << 16);                              // data_size = 8 bytes
  g1[1] = (int)(1024u << 16);                           // tensor_dim0 = 1024 (lo16)
  g1[2] = (int)(1u << 16);                              // dim0 hi=0 | tensor_dim1 = 1
  g1[3] = (int)(1024u << 16);                           // dim1 hi=0 | tile_dim0 = 1024
  g1[4] = 1;                                            // tile_dim1 = 1, tile_dim2 = 0
  g1[5] = 1024;                                         // tensor_dim0_stride = 1024
  g1[6] = 0;
  g1[7] = 0;
  i32x4_t z4 = {0, 0, 0, 0};
  i32x8_t z8 = {0, 0, 0, 0, 0, 0, 0, 0};
  __builtin_amdgcn_tensor_load_to_lds(g0, g1, z4, z4, z8, 0);
}

// ---------------------------------------------------------------------------
// Kernel 1: pre-split W (56x768 fp32) into bf16 hi/lo fragments in a contiguous-k
// V_WMMA_F32_16X16X32_BF16 B-matrix VGPR layout, one 8KB slab per k-tile:
//   lane 0-15  -> K = e (element 0..15), N = lane
//   lane 16-31 -> K = 16 + e,           N = lane - 16
// ---------------------------------------------------------------------------
__global__ void prep_w_kernel(const float* __restrict__ W,
                              __bf16* __restrict__ wfr) {
  int tid = blockIdx.x * blockDim.x + threadIdx.x;
  if (tid >= WFR_ELEMS / 2) return;        // one thread per (hi,lo) pair
  int e    = tid & 15;
  int lane = (tid >> 4) & 31;
  int nt4  = (tid >> 9) & 3;
  int kt   = tid >> 11;
  int d = kt * 32 + (lane & 16) + e;       // K index within full D
  int n = nt4 * 16 + (lane & 15);          // output column
  float v = (n < KC_) ? W[n * D_ + d] : 0.0f;
  __bf16 h = (__bf16)v;
  size_t base = (size_t)kt * SLAB_ELEMS + ((nt4 * 32 + lane) * 16 + e);
  wfr[base]        = h;                             // hi half of slab
  wfr[base + 2048] = (__bf16)(v - (float)h);        // lo half of slab
}

// ---------------------------------------------------------------------------
// Kernel 2: emission GEMM via bf16 WMMA, 3-term hi/lo split (~fp32 accuracy).
// 8 waves / block, each wave owns one 16-row M tile; C = 16x64 (4 N-tiles).
// W fragments staged global->LDS with the Tensor Data Mover (double buffer);
// A fragments loaded straight from global in contiguous-k order (same as B):
//   lane 0-15:  M = lane,     K = koff + {0..15}, koff = 0
//   lane 16-31: M = lane-16,  same pattern with koff = 16
// ---------------------------------------------------------------------------
__global__ void __launch_bounds__(256)
emission_gemm_kernel(const float* __restrict__ X,      // (B*S, D)
                     const __bf16* __restrict__ wfr,   // k-tile slabs
                     const float* __restrict__ bias,   // (56,)
                     float* __restrict__ em) {         // (B, NT, S, T)
  __shared__ __bf16 ldsw[2 * SLAB_ELEMS];              // 2 x 8KB double buffer

  const int lane = threadIdx.x & 31;
  const int wave = threadIdx.x >> 5;
  const bool issuer = (wave == 0);                     // wave-uniform
  const int mtile = blockIdx.x * 8 + wave;             // 0 .. 4095
  const int R = mtile * 16;

  const int rowInTile = lane & 15;
  const int koff = (lane & 16) ? 16 : 0;
  const float* aptr = X + (size_t)(R + rowInTile) * D_;

  v8f c[4];
  #pragma unroll
  for (int i = 0; i < 4; ++i) c[i] = (v8f){0.f,0.f,0.f,0.f,0.f,0.f,0.f,0.f};

  // Prologue: DMA k-tile 0 slab into buffer 0.
  if (issuer)
    tdm_load_slab(wfr, (unsigned)(uintptr_t)&ldsw[0]);

  for (int kt = 0; kt < KTILES_; ++kt) {
    const int buf = kt & 1;

    // Wait for this k-tile's TDM transfer, then make it visible block-wide.
    if (issuer) __builtin_amdgcn_s_wait_tensorcnt((short)0);
    __syncthreads();

    // Overlap: kick off the next slab into the other buffer (its previous
    // readers all consumed their ds loads before signaling the barrier).
    if (issuer && (kt + 1 < KTILES_))
      tdm_load_slab(wfr + (size_t)(kt + 1) * SLAB_ELEMS,
                    (unsigned)(uintptr_t)&ldsw[(buf ^ 1) * SLAB_ELEMS]);

    // ---- A fragment: 4 x float4 from global, convert to bf16 hi/lo ----
    const int kb = kt * 32 + koff;
    if (kt + 1 < KTILES_)
      __builtin_prefetch(aptr + kb + 32, 0, 1);        // next k-tile of A

    const float4 f0 = *(const float4*)(aptr + kb + 0);
    const float4 f1 = *(const float4*)(aptr + kb + 4);
    const float4 f2 = *(const float4*)(aptr + kb + 8);
    const float4 f3 = *(const float4*)(aptr + kb + 12);

    float av[16] = {f0.x,f0.y,f0.z,f0.w, f1.x,f1.y,f1.z,f1.w,
                    f2.x,f2.y,f2.z,f2.w, f3.x,f3.y,f3.z,f3.w};
    v16bf ahi, alo;
    #pragma unroll
    for (int e = 0; e < 16; ++e) {
      __bf16 h = (__bf16)av[e];
      ahi[e] = h;
      alo[e] = (__bf16)(av[e] - (float)h);
    }

    // ---- B fragments from LDS (ds_load_b128), 3-term split WMMA ----
    const __bf16* slab = &ldsw[buf * SLAB_ELEMS];
    #pragma unroll
    for (int nt4 = 0; nt4 < 4; ++nt4) {
      const __bf16* fh = slab + (nt4 * 32 + lane) * 16;
      v16bf bhi = *(const v16bf*)(fh);
      v16bf blo = *(const v16bf*)(fh + 2048);
      c[nt4] = __builtin_amdgcn_wmma_f32_16x16x32_bf16(
          false, ahi, false, bhi, (short)0, c[nt4], false, false);
      c[nt4] = __builtin_amdgcn_wmma_f32_16x16x32_bf16(
          false, ahi, false, blo, (short)0, c[nt4], false, false);
      c[nt4] = __builtin_amdgcn_wmma_f32_16x16x32_bf16(
          false, alo, false, bhi, (short)0, c[nt4], false, false);
    }
  }

  // Scatter C (+bias) into emission (B, NT, S, T).
  // C layout: VGPR j -> M = j (lanes 0-15) / j+8 (lanes 16-31); N = lane%16.
  const int nbase  = lane & 15;
  const int rowadd = (lane & 16) ? 8 : 0;
  #pragma unroll
  for (int nt4 = 0; nt4 < 4; ++nt4) {
    const int n = nt4 * 16 + nbase;
    if (n < KC_) {
      const float bn = bias[n];
      const int nt = n >> 2, t = n & 3;
      #pragma unroll
      for (int j = 0; j < 8; ++j) {
        const int g = R + j + rowadd;       // global row in (B*S)
        const int s = g & (S_ - 1);
        const int bb = g >> 10;
        em[(((size_t)bb * NT_ + nt) * S_ + s) * T_ + t] = c[nt4][j] + bn;
      }
    }
  }
}

// ---------------------------------------------------------------------------
// Kernel 3: Viterbi (forward with 2-bit-packed backpointers + fused
// backtrack). One thread per sequence n in [0, 896).
// ---------------------------------------------------------------------------
__global__ void viterbi_kernel(const float* __restrict__ em,        // (B,NT,S,T)
                               const unsigned char* __restrict__ mask, // (B,S)
                               const float* __restrict__ start_trans,  // (4,)
                               const float* __restrict__ end_trans,    // (4,)
                               const float* __restrict__ trans,        // (4,4)
                               unsigned char* __restrict__ bp,  // (896, 1023)
                               float* __restrict__ tags) {      // (B,NT,S)
  const int n = blockIdx.x * blockDim.x + threadIdx.x;
  if (n >= NSEQ_) return;
  const int bb = n / NT_;

  // sequence length from mask
  int len = 0;
  for (int s = 0; s < S_; ++s) len += (mask[(size_t)bb * S_ + s] != 0);

  float tr[4][4];
  #pragma unroll
  for (int p = 0; p < 4; ++p)
    #pragma unroll
    for (int t = 0; t < 4; ++t) tr[p][t] = trans[p * 4 + t];

  const float4* E = (const float4*)(em + (size_t)n * S_ * T_);
  unsigned char* bpn = bp + (size_t)n * (S_ - 1);

  float4 e0 = E[0];
  float sc[4] = { start_trans[0] + e0.x, start_trans[1] + e0.y,
                  start_trans[2] + e0.z, start_trans[3] + e0.w };

  for (int s = 1; s < S_; ++s) {
    const float4 e = E[s];
    const float ev[4] = {e.x, e.y, e.z, e.w};
    float ns[4];
    unsigned byte = 0;
    #pragma unroll
    for (int t = 0; t < 4; ++t) {
      float best = sc[0] + tr[0][t];
      int bi = 0;
      #pragma unroll
      for (int p = 1; p < 4; ++p) {
        float cand = sc[p] + tr[p][t];
        if (cand > best) { best = cand; bi = p; }   // first-max tie rule
      }
      ns[t] = best + ev[t];
      byte |= (unsigned)bi << (2 * t);
    }
    bpn[s - 1] = (unsigned char)byte;
    if (mask[(size_t)bb * S_ + s]) {
      #pragma unroll
      for (int t = 0; t < 4; ++t) sc[t] = ns[t];
    }
  }

  // final scores + argmax (first max)
  float fbest = sc[0] + end_trans[0];
  int best_last = 0;
  #pragma unroll
  for (int t = 1; t < 4; ++t) {
    float f = sc[t] + end_trans[t];
    if (f > fbest) { fbest = f; best_last = t; }
  }

  // backtrack
  float* out = tags + (size_t)n * S_;
  int cur = best_last;
  for (int s = S_ - 1; s >= 0; --s) {
    float outv;
    if (s >= len) {
      outv = 0.0f;
    } else {
      int nx = (s == len - 1) ? best_last
                              : (int)((bpn[s] >> (2 * cur)) & 3u);
      cur = nx;
      outv = (float)nx;
    }
    out[s] = outv;
  }
}

// ---------------------------------------------------------------------------
extern "C" void kernel_launch(void* const* d_in, const int* in_sizes, int n_in,
                              void* d_out, int out_size, void* d_ws, size_t ws_size,
                              hipStream_t stream) {
  const float*         text_vec    = (const float*)d_in[0];        // (B,S,D)
  const unsigned char* mask        = (const unsigned char*)d_in[1];// (B,S) bool
  const float*         W           = (const float*)d_in[2];        // (56,768)
  const float*         b           = (const float*)d_in[3];        // (56,)
  const float*         start_trans = (const float*)d_in[4];        // (4,)
  const float*         end_trans   = (const float*)d_in[5];        // (4,)
  const float*         trans       = (const float*)d_in[6];        // (4,4)

  float* em   = (float*)d_out;                              // (B,NT,S,T)
  float* tags = (float*)d_out + (size_t)B_ * NT_ * S_ * T_; // (B,NT,S)

  __bf16* wfr = (__bf16*)d_ws;                              // k-tile slabs
  unsigned char* bp = (unsigned char*)d_ws + BP_OFF_BYTES;  // (896,1023)

  // 1) W -> bf16 hi/lo fragment slabs (WMMA B layout, one slab per k-tile)
  prep_w_kernel<<<(WFR_ELEMS / 2 + 255) / 256, 256, 0, stream>>>(W, wfr);

  // 2) emission GEMM: 4096 M-tiles, 8 waves (tiles) per 256-thread block,
  //    W slabs staged via Tensor Data Mover into double-buffered LDS
  emission_gemm_kernel<<<MROWS_ / 16 / 8, 256, 0, stream>>>(
      text_vec, wfr, b, em);

  // 3) Viterbi decode, one thread per (b, nt) sequence
  viterbi_kernel<<<(NSEQ_ + 127) / 128, 128, 0, stream>>>(
      em, mask, start_trans, end_trans, trans, bp, tags);
}
